// CPPN_47330539602764
// MI455X (gfx1250) — matrix operand order repaired and checked by reading, not compiled, not run on hardware
//
#include <hip/hip_runtime.h>

typedef __attribute__((ext_vector_type(16))) __bf16 v16bf;
typedef __attribute__((ext_vector_type(8)))  __bf16 v8bf;
typedef __attribute__((ext_vector_type(8)))  float  v8f;
typedef __attribute__((ext_vector_type(4)))  int    v4i;

#define EPS 1e-5f

#if defined(__has_builtin)
#if __has_builtin(__builtin_amdgcn_global_load_async_to_lds_b128)
#define HAVE_ASYNC_LDS 1
#endif
#endif

static __device__ __forceinline__ float leakyf(float x) { return x >= 0.f ? x : 0.2f * x; }
static __device__ __forceinline__ __bf16 f2bf(float f) { return (__bf16)f; }

// Stage 32 bytes global -> LDS (async path if available, else via VGPRs).
static __device__ __forceinline__ void stage32(const __bf16* s, __bf16* d)
{
#ifdef HAVE_ASYNC_LDS
  __builtin_amdgcn_global_load_async_to_lds_b128(
      (v4i*)s, (__attribute__((address_space(3))) v4i*)d, 0, 0);
  __builtin_amdgcn_global_load_async_to_lds_b128(
      (v4i*)(s + 8), (__attribute__((address_space(3))) v4i*)(d + 8), 0, 0);
#else
  *(v8bf*)d       = *(const v8bf*)s;
  *(v8bf*)(d + 8) = *(const v8bf*)(s + 8);
#endif
}

static __device__ __forceinline__ void wait_async()
{
#ifdef HAVE_ASYNC_LDS
#if __has_builtin(__builtin_amdgcn_s_wait_asynccnt)
  __builtin_amdgcn_s_wait_asynccnt(0);
#else
  asm volatile("s_wait_asynccnt 0x0" ::: "memory");
#endif
#endif
}

// ---------------------------------------------------------------------------
// Generic bf16 WMMA GEMM:  C[M,N] = A[M,Kp] * Bt[N,Kp]^T (+bias[n])
// A, Bt row-major bf16; Kp multiple of 32.
// One wave -> one 16x32 C tile (2 N-tiles, A reused). Block = 8 waves = 32x128.
// ---------------------------------------------------------------------------
__global__ __launch_bounds__(256) void k_gemm_bt(
    const __bf16* __restrict__ A, const __bf16* __restrict__ Bt,
    const float* __restrict__ bias, float* __restrict__ C,
    int M, int N, int Kp, int ldc)
{
  int wid  = threadIdx.x >> 5;
  int lane = threadIdx.x & 31;
  int half = lane >> 4;
  int l    = lane & 15;
  int m0 = blockIdx.y * 32 + (wid >> 2) * 16;
  int n0 = blockIdx.x * 128 + (wid & 3) * 32;
  if (m0 >= M || n0 >= N) return;

  const __bf16* arow  = A  + (size_t)(m0 + l) * Kp;
  const __bf16* brow0 = Bt + (size_t)(n0 + l) * Kp + half * 16;
  const __bf16* brow1 = Bt + (size_t)(n0 + 16 + l) * Kp + half * 16;
  v8f acc0 = {}, acc1 = {};
  for (int k0 = 0; k0 < Kp; k0 += 32) {
    union { v16bf v; v8bf h[2]; } a;
    a.h[0] = *(const v8bf*)(arow + k0 + half * 16);      // K = h*16 .. h*16+7
    a.h[1] = *(const v8bf*)(arow + k0 + half * 16 + 8);  // K = h*16+8 .. h*16+15
    v16bf b0 = *(const v16bf*)(brow0 + k0);              // K = h*16 .. h*16+15
    v16bf b1 = *(const v16bf*)(brow1 + k0);
    acc0 = __builtin_amdgcn_wmma_f32_16x16x32_bf16(false, a.v, false, b0,
                                                   (short)0, acc0, false, false);
    acc1 = __builtin_amdgcn_wmma_f32_16x16x32_bf16(false, a.v, false, b1,
                                                   (short)0, acc1, false, false);
  }
  int na = n0 + l;
  int nb = n0 + 16 + l;
  float bva = (bias && na < N) ? bias[na] : 0.f;
  float bvb = (bias && nb < N) ? bias[nb] : 0.f;
#pragma unroll
  for (int r = 0; r < 8; ++r) {
    int m = m0 + half * 8 + r;
    if (m < M) {
      if (na < N) C[(size_t)m * ldc + na] = acc0[r] + bva;
      if (nb < N) C[(size_t)m * ldc + nb] = acc1[r] + bvb;
    }
  }
}

// ---------------------------------------------------------------------------
// Fused edge-MLP first layer: G[p*200+q, n] = sum_k (x[p,k]-x[q,k])^2 * W1[k,n] + b1[n]
// M=40000, N=128, K=2048. A generated on the fly once per wave per k-step and
// reused across all 8 N-tiles; the 128x32 B-tile of W1^T is staged into LDS per
// block (double-buffered, async-load-to-LDS when available, shared by 8 waves).
// ---------------------------------------------------------------------------
__global__ __launch_bounds__(256) void k_edge_gemm(
    const float* __restrict__ X, const __bf16* __restrict__ W1t,
    const float* __restrict__ b1, float* __restrict__ G)
{
  __shared__ __bf16 lb[2][128 * 40];  // row stride 40 (80B) -> conflict-free ds reads
  int tid  = threadIdx.x;
  int wid  = tid >> 5;
  int lane = tid & 31;
  int half = lane >> 4;
  int l    = lane & 15;
  int m0 = blockIdx.x * 128 + wid * 16;
  bool active = m0 < 40000;
  int m = (active ? m0 : 0) + l;
  int p = m / 200, q = m % 200;
  const float* xp = X + (size_t)p * 2048;
  const float* xq = X + (size_t)q * 2048;

  int srow = tid >> 1;          // 0..127
  int skc  = (tid & 1) * 16;    // 0 or 16
  const __bf16* sbase = W1t + (size_t)srow * 2048 + skc;

  v8f acc[8];
#pragma unroll
  for (int t = 0; t < 8; ++t) acc[t] = (v8f){};

  // prologue: stage k-tile 0 into buffer 0
  stage32(sbase, &lb[0][srow * 40 + skc]);
  wait_async();
  __syncthreads();

  for (int kt = 0; kt < 64; ++kt) {
    int k0  = kt * 32;
    int cur = kt & 1;
    if (kt + 1 < 64)  // stage next k-tile into the other buffer
      stage32(sbase + (size_t)(kt + 1) * 32, &lb[1 - cur][srow * 40 + skc]);

    union { v16bf v; v8bf h[2]; } a;
#pragma unroll
    for (int i = 0; i < 8; ++i) {
      float d0 = xp[k0 + half * 16 + i]      - xq[k0 + half * 16 + i];
      float d1 = xp[k0 + half * 16 + 8 + i] - xq[k0 + half * 16 + 8 + i];
      a.h[0][i] = f2bf(d0 * d0);
      a.h[1][i] = f2bf(d1 * d1);
    }
    __builtin_prefetch(xp + k0 + 256, 0, 0);
    __builtin_prefetch(xq + k0 + 256, 0, 0);
#pragma unroll
    for (int t = 0; t < 8; ++t) {
      union { v16bf v; v8bf h[2]; } b;
      const __bf16* bp = &lb[cur][(t * 16 + l) * 40 + half * 16];
      b.h[0] = *(const v8bf*)bp;
      b.h[1] = *(const v8bf*)(bp + 8);
      acc[t] = __builtin_amdgcn_wmma_f32_16x16x32_bf16(false, a.v, false, b.v,
                                                       (short)0, acc[t], false, false);
    }
    wait_async();
    __syncthreads();
  }

  if (active) {
#pragma unroll
    for (int t = 0; t < 8; ++t) {
      int n = t * 16 + l;
      float bv = b1[n];
#pragma unroll
      for (int r = 0; r < 8; ++r)
        G[(size_t)(m0 + half * 8 + r) * 128 + n] = acc[t][r] + bv;
    }
  }
}

// ---------------------------------------------------------------------------
// Elementwise / reduction kernels
// ---------------------------------------------------------------------------
__global__ void k_offs(const float* __restrict__ attr, const float* __restrict__ cen,
                       __bf16* __restrict__ offs)
{
  size_t idx = (size_t)blockIdx.x * blockDim.x + threadIdx.x;
  const size_t total = (size_t)3 * 208 * 320;
  if (idx >= total) return;
  int k = (int)(idx / (208 * 320));
  int r = (int)(idx % (208 * 320));
  int m = r / 320, s = r % 320;
  float v = (m < 200 && s < 312) ? (attr[m * 312 + s] - cen[k * 312 + s]) : 0.f;
  offs[idx] = f2bf(v);
}

// out[c,r'] = in[r',c] for r'<R else 0 ; out is [Cc, Kp] row-major
__global__ void k_transpose(const float* __restrict__ in, __bf16* __restrict__ out,
                            int R, int Cc, int Kp)
{
  size_t idx = (size_t)blockIdx.x * blockDim.x + threadIdx.x;
  size_t total = (size_t)Cc * Kp;
  if (idx >= total) return;
  int c = (int)(idx / Kp);
  int r = (int)(idx % Kp);
  out[idx] = (r < R) ? f2bf(in[(size_t)r * Cc + c]) : f2bf(0.f);
}

// out[m,c] = in[m,c] for c<Cc else 0 ; out is [M, Kp]
__global__ void k_convert(const float* __restrict__ in, __bf16* __restrict__ out,
                          int M, int Cc, int Kp)
{
  size_t idx = (size_t)blockIdx.x * blockDim.x + threadIdx.x;
  size_t total = (size_t)M * Kp;
  if (idx >= total) return;
  int c = (int)(idx % Kp);
  size_t m = idx / Kp;
  out[idx] = (c < Cc) ? f2bf(in[m * Cc + c]) : f2bf(0.f);
}

__global__ void k_combine(const float* __restrict__ resp, const float* __restrict__ eb,
                          float* __restrict__ spF, __bf16* __restrict__ spA)
{
  size_t idx = (size_t)blockIdx.x * blockDim.x + threadIdx.x;
  if (idx >= (size_t)200 * 2048) return;
  int d = (int)(idx % 2048);
  float s = 0.f;
#pragma unroll
  for (int k = 0; k < 3; ++k)
    s += fmaxf(resp[(size_t)k * 200 * 2048 + idx] + eb[k * 2048 + d], 0.f);
  spF[idx] = s;
  spA[idx] = f2bf(s);
}

__global__ void k_colstats(const float* __restrict__ h, float* __restrict__ mu,
                           float* __restrict__ var, int M, int D)
{
  int d = blockIdx.x * blockDim.x + threadIdx.x;
  if (d >= D) return;
  float s = 0.f, sq = 0.f;
  for (int m = 0; m < M; ++m) {
    float x = h[(size_t)m * D + d];
    s += x; sq += x * x;
  }
  float mm = s / M;
  mu[d] = mm;
  var[d] = sq / M - mm * mm;
}

__global__ void k_bn_apply(const float* __restrict__ h, const float* __restrict__ mu,
                           const float* __restrict__ var, const float* __restrict__ g,
                           const float* __restrict__ b, __bf16* __restrict__ outA,
                           int M, int D)
{
  size_t idx = (size_t)blockIdx.x * blockDim.x + threadIdx.x;
  if (idx >= (size_t)M * D) return;
  int d = (int)(idx % D);
  float y = (h[idx] - mu[d]) * rsqrtf(var[d] + EPS) * g[d] + b[d];
  outA[idx] = f2bf(leakyf(y));
}

// per-row instance-norm; mode 0: leaky(y) ; mode 1: relu(y)+res
__global__ void k_rownorm(const float* __restrict__ in, const float* __restrict__ res,
                          float* __restrict__ outF, __bf16* __restrict__ outA,
                          int D, int mode)
{
  int row = blockIdx.x, tid = threadIdx.x;
  __shared__ float s1[256], s2[256];
  float sum = 0.f, sq = 0.f;
  for (int d = tid; d < D; d += 256) {
    float x = in[(size_t)row * D + d];
    sum += x; sq += x * x;
  }
  s1[tid] = sum; s2[tid] = sq; __syncthreads();
  for (int s = 128; s > 0; s >>= 1) {
    if (tid < s) { s1[tid] += s1[tid + s]; s2[tid] += s2[tid + s]; }
    __syncthreads();
  }
  float mean = s1[0] / D;
  float v    = s2[0] / D - mean * mean;
  float inv  = rsqrtf(v + EPS);
  for (int d = tid; d < D; d += 256) {
    float y = (in[(size_t)row * D + d] - mean) * inv;
    float o = mode ? (fmaxf(y, 0.f) + res[(size_t)row * D + d]) : leakyf(y);
    outF[(size_t)row * D + d] = o;
    if (outA) outA[(size_t)row * D + d] = f2bf(o);
  }
}

__global__ void k_l2rows(const float* __restrict__ in, __bf16* __restrict__ outA,
                         int D, int Kp)
{
  int row = blockIdx.x, tid = threadIdx.x;
  __shared__ float s1[256];
  float sq = 0.f;
  for (int d = tid; d < D; d += 256) { float x = in[(size_t)row * D + d]; sq += x * x; }
  s1[tid] = sq; __syncthreads();
  for (int s = 128; s > 0; s >>= 1) { if (tid < s) s1[tid] += s1[tid + s]; __syncthreads(); }
  float inv = 1.f / fmaxf(sqrtf(s1[0]), 1e-8f);
  for (int c = tid; c < Kp; c += 256)
    outA[(size_t)row * Kp + c] = (c < D) ? f2bf(in[(size_t)row * D + c] * inv) : f2bf(0.f);
}

__global__ void k_rowsoftmax(const float* __restrict__ in, float* __restrict__ outF,
                             __bf16* __restrict__ outA, int Nc, int Kp, float invT)
{
  int row = blockIdx.x, tid = threadIdx.x;
  __shared__ float s1[256];
  float mx = -3.4e38f;
  for (int c = tid; c < Nc; c += 256) mx = fmaxf(mx, in[(size_t)row * Nc + c]);
  s1[tid] = mx; __syncthreads();
  for (int s = 128; s > 0; s >>= 1) { if (tid < s) s1[tid] = fmaxf(s1[tid], s1[tid + s]); __syncthreads(); }
  mx = s1[0]; __syncthreads();
  float sum = 0.f;
  for (int c = tid; c < Nc; c += 256) sum += expf((in[(size_t)row * Nc + c] - mx) * invT);
  s1[tid] = sum; __syncthreads();
  for (int s = 128; s > 0; s >>= 1) { if (tid < s) s1[tid] += s1[tid + s]; __syncthreads(); }
  float inv = 1.f / s1[0];
  for (int c = tid; c < Kp; c += 256) {
    if (c < Nc) {
      float v = expf((in[(size_t)row * Nc + c] - mx) * invT) * inv;
      outF[(size_t)row * Nc + c] = v;
      outA[(size_t)row * Kp + c] = f2bf(v);
    } else {
      outA[(size_t)row * Kp + c] = f2bf(0.f);
    }
  }
}

__global__ void k_add_conv(const float* __restrict__ a, const float* __restrict__ b,
                           float* __restrict__ oF, __bf16* __restrict__ oA, size_t n)
{
  size_t idx = (size_t)blockIdx.x * blockDim.x + threadIdx.x;
  if (idx >= n) return;
  float v = a[idx] + b[idx];
  oF[idx] = v;
  oA[idx] = f2bf(v);
}

// Fused edge-MLP tail: per row (40000): instnorm(H=128) -> relu -> dot W2 -> tanh -> *(last+1e-8)
__global__ void k_edge_row(const float* __restrict__ G, const float* __restrict__ W2,
                           const float* __restrict__ b2, const float* __restrict__ last,
                           float* __restrict__ pre)
{
  int row = blockIdx.x, tid = threadIdx.x; // 128 threads
  __shared__ float s1[128], s2[128];
  float g = G[(size_t)row * 128 + tid];
  s1[tid] = g; s2[tid] = g * g; __syncthreads();
  for (int s = 64; s > 0; s >>= 1) {
    if (tid < s) { s1[tid] += s1[tid + s]; s2[tid] += s2[tid + s]; }
    __syncthreads();
  }
  float mean = s1[0] / 128.f;
  float var  = s2[0] / 128.f - mean * mean;
  __syncthreads();
  float h = fmaxf((g - mean) * rsqrtf(var + EPS), 0.f);
  s1[tid] = h * W2[tid];
  __syncthreads();
  for (int s = 64; s > 0; s >>= 1) { if (tid < s) s1[tid] += s1[tid + s]; __syncthreads(); }
  if (tid == 0) {
    float cur = tanhf(s1[0] + b2[0]);
    pre[row] = cur * (last[row] + 1e-8f);
  }
}

__global__ void k_fusion(const float* __restrict__ Vv, const float* __restrict__ Vs,
                         const float* __restrict__ u, const float* __restrict__ vprot,
                         const float* __restrict__ sprot, __bf16* __restrict__ protoA)
{
  int row = blockIdx.x, tid = threadIdx.x;
  __shared__ float r0[256], r1[256];
  __shared__ float a0s, a1s;
  float sv = 0.f, ss = 0.f;
  for (int d = tid; d < 2048; d += 256) {
    float uu = u[d];
    sv += tanhf(Vv[(size_t)row * 2048 + d]) * uu;
    ss += tanhf(Vs[(size_t)row * 2048 + d]) * uu;
  }
  r0[tid] = sv; r1[tid] = ss; __syncthreads();
  for (int s = 128; s > 0; s >>= 1) {
    if (tid < s) { r0[tid] += r0[tid + s]; r1[tid] += r1[tid + s]; }
    __syncthreads();
  }
  if (tid == 0) {
    float l0 = r0[0] * 100.f + 1e-8f, l1 = r1[0] * 100.f + 1e-8f;
    float mx = fmaxf(l0, l1);
    float e0 = expf(l0 - mx), e1 = expf(l1 - mx);
    a0s = e0 / (e0 + e1); a1s = e1 / (e0 + e1);
  }
  __syncthreads();
  float a0 = a0s, a1 = a1s;
  for (int d = tid; d < 2048; d += 256)
    protoA[(size_t)row * 2048 + d] =
        f2bf(a0 * vprot[(size_t)row * 2048 + d] + a1 * sprot[(size_t)row * 2048 + d]);
}

// ---------------------------------------------------------------------------
extern "C" void kernel_launch(void* const* d_in, const int* in_sizes, int n_in,
                              void* d_out, int out_size, void* d_ws, size_t ws_size,
                              hipStream_t stream)
{
  (void)in_sizes; (void)n_in; (void)out_size; (void)ws_size;
  const float* img  = (const float*)d_in[0];
  const float* attr = (const float*)d_in[1];
  const float* cen  = (const float*)d_in[2];
  const float* eW   = (const float*)d_in[3];
  const float* eb   = (const float*)d_in[4];
  const float* W1   = (const float*)d_in[5];
  const float* b1   = (const float*)d_in[6];
  const float* bng  = (const float*)d_in[7];
  const float* bnb  = (const float*)d_in[8];
  const float* W2   = (const float*)d_in[9];
  const float* b2   = (const float*)d_in[10];
  const float* vnW  = (const float*)d_in[11];
  const float* vnb  = (const float*)d_in[12];
  const float* veW1 = (const float*)d_in[13];
  const float* veb1 = (const float*)d_in[14];
  const float* veW2 = (const float*)d_in[15];
  const float* veb2 = (const float*)d_in[16];
  const float* snW  = (const float*)d_in[17];
  const float* snb  = (const float*)d_in[18];
  // d_in[19..22] (se_*) are dead: the updated semantic edge is never consumed.
  const float* fusW = (const float*)d_in[23];
  const float* fusu = (const float*)d_in[24];
  float* out = (float*)d_out;

  const int Nn = 200, NP = 208, NB = 224, SP = 320, D = 2048, KE = 224, Bb = 2048;

  char* p = (char*)d_ws;
  auto carve = [&](size_t bytes) -> void* {
    void* r = (void*)p;
    p += (bytes + 255) & ~(size_t)255;
    return r;
  };
  // bf16 arenas
  __bf16* offs    = (__bf16*)carve((size_t)3 * NP * SP * 2);
  __bf16* eWt     = (__bf16*)carve((size_t)3 * D * SP * 2);
  __bf16* W1t     = (__bf16*)carve((size_t)D * D * 2);
  __bf16* W2t     = (__bf16*)carve((size_t)D * D * 2);
  __bf16* vnWt    = (__bf16*)carve((size_t)D * D * 2);
  __bf16* snWt    = (__bf16*)carve((size_t)D * D * 2);
  __bf16* fusWt   = (__bf16*)carve((size_t)D * D * 2);
  __bf16* veW1t   = (__bf16*)carve((size_t)128 * D * 2);
  __bf16* spA     = (__bf16*)carve((size_t)NP * D * 2);
  __bf16* h1A     = (__bf16*)carve((size_t)NP * D * 2);
  __bf16* xnvA    = (__bf16*)carve((size_t)NB * D * 2);   // Bt rows padded to 224
  __bf16* xnaA    = (__bf16*)carve((size_t)NB * SP * 2);  // Bt rows padded to 224
  __bf16* edgeV0A = (__bf16*)carve((size_t)NP * KE * 2);
  __bf16* edgeS0A = (__bf16*)carve((size_t)NP * KE * 2);
  __bf16* edgeV1A = (__bf16*)carve((size_t)NP * KE * 2);
  __bf16* spT     = (__bf16*)carve((size_t)D * KE * 2);
  __bf16* pvpT    = (__bf16*)carve((size_t)D * KE * 2);
  __bf16* sumvpA  = (__bf16*)carve((size_t)NP * D * 2);
  __bf16* sumspA  = (__bf16*)carve((size_t)NP * D * 2);
  __bf16* vprotA  = (__bf16*)carve((size_t)NP * D * 2);
  __bf16* sprotA  = (__bf16*)carve((size_t)NP * D * 2);
  __bf16* imgA    = (__bf16*)carve((size_t)Bb * D * 2);
  __bf16* protoA  = (__bf16*)carve((size_t)NB * D * 2);   // Bt rows padded to 224
  // f32 arenas
  float* resp    = (float*)carve((size_t)3 * Nn * D * 4);
  float* sprotoF = (float*)carve((size_t)Nn * D * 4);
  float* h1F     = (float*)carve((size_t)Nn * D * 4);
  float* mu      = (float*)carve((size_t)D * 4);
  float* var     = (float*)carve((size_t)D * 4);
  float* h2F     = (float*)carve((size_t)Nn * D * 4);
  float* pvpF    = (float*)carve((size_t)Nn * D * 4);
  float* simV    = (float*)carve((size_t)Nn * Nn * 4);
  float* edgeVF  = (float*)carve((size_t)Nn * Nn * 4);
  float* simS    = (float*)carve((size_t)Nn * Nn * 4);
  float* edgeSF  = (float*)carve((size_t)Nn * Nn * 4);
  float* preE    = (float*)carve((size_t)Nn * Nn * 4);
  float* edgeV1F = (float*)carve((size_t)Nn * Nn * 4);
  float* vpF     = (float*)carve((size_t)Nn * D * 4);
  float* exvpF   = (float*)carve((size_t)Nn * D * 4);
  float* sumvpF  = (float*)carve((size_t)Nn * D * 4);
  float* tF      = (float*)carve((size_t)Nn * D * 4);
  float* vprotF  = (float*)carve((size_t)Nn * D * 4);
  float* spFn    = (float*)carve((size_t)Nn * D * 4);
  float* exspF   = (float*)carve((size_t)Nn * D * 4);
  float* sumspF  = (float*)carve((size_t)Nn * D * 4);
  float* t2F     = (float*)carve((size_t)Nn * D * 4);
  float* sprotF  = (float*)carve((size_t)Nn * D * 4);
  float* G       = (float*)carve((size_t)40000 * 128 * 4);
  float* Vv      = (float*)carve((size_t)Nn * D * 4);
  float* Vs      = (float*)carve((size_t)Nn * D * 4);

  auto blocks = [](size_t total) { return (unsigned)((total + 255) / 256); };
  auto gemm = [&](const __bf16* A, const __bf16* Bt, const float* bias, float* C,
                  int M, int N_, int Kp, int ldc) {
    dim3 g((N_ + 127) / 128, (M + 31) / 32);
    k_gemm_bt<<<g, 256, 0, stream>>>(A, Bt, bias, C, M, N_, Kp, ldc);
  };

  // ---- operand preparation (bf16 conversions / transposes) ----
  k_offs<<<blocks((size_t)3 * NP * SP), 256, 0, stream>>>(attr, cen, offs);
  for (int k = 0; k < 3; ++k)
    k_transpose<<<blocks((size_t)D * SP), 256, 0, stream>>>(
        eW + (size_t)k * 312 * D, eWt + (size_t)k * D * SP, 312, D, SP);
  k_transpose<<<blocks((size_t)D * D), 256, 0, stream>>>(W1,  W1t,  D, D, D);
  k_transpose<<<blocks((size_t)D * D), 256, 0, stream>>>(W2,  W2t,  D, D, D);
  k_transpose<<<blocks((size_t)D * D), 256, 0, stream>>>(vnW, vnWt, D, D, D);
  k_transpose<<<blocks((size_t)D * D), 256, 0, stream>>>(snW, snWt, D, D, D);
  k_transpose<<<blocks((size_t)D * D), 256, 0, stream>>>(fusW, fusWt, D, D, D);
  k_transpose<<<blocks((size_t)128 * D), 256, 0, stream>>>(veW1, veW1t, D, 128, D);
  k_convert<<<blocks((size_t)Bb * D), 256, 0, stream>>>(img, imgA, Bb, D, D);

  // ---- cooperation module ----
  for (int k = 0; k < 3; ++k)
    gemm(offs + (size_t)k * NP * SP, eWt + (size_t)k * D * SP, nullptr,
         resp + (size_t)k * Nn * D, Nn, D, SP, D);
  k_combine<<<blocks((size_t)Nn * D), 256, 0, stream>>>(resp, eb, sprotoF, spA);

  // ---- semantic2visual ----
  gemm(spA, W1t, b1, h1F, Nn, D, D, D);
  k_colstats<<<D / 256, 256, 0, stream>>>(h1F, mu, var, Nn, D);
  k_bn_apply<<<blocks((size_t)Nn * D), 256, 0, stream>>>(h1F, mu, var, bng, bnb, h1A, Nn, D);
  gemm(h1A, W2t, b2, h2F, Nn, D, D, D);
  k_rownorm<<<Nn, 256, 0, stream>>>(h2F, nullptr, pvpF, nullptr, D, 0);

  // ---- cosine edges ----
  k_l2rows<<<Nn, 256, 0, stream>>>(pvpF, xnvA, D, D);
  gemm(xnvA, xnvA, nullptr, simV, Nn, Nn, D, Nn);
  k_rowsoftmax<<<Nn, 256, 0, stream>>>(simV, edgeVF, edgeV0A, Nn, KE, 100.f);
  k_l2rows<<<Nn, 256, 0, stream>>>(attr, xnaA, 312, SP);
  gemm(xnaA, xnaA, nullptr, simS, Nn, Nn, SP, Nn);
  k_rowsoftmax<<<Nn, 256, 0, stream>>>(simS, edgeSF, edgeS0A, Nn, KE, 100.f);

  // ---- UpdateVisualNode ----
  k_transpose<<<blocks((size_t)D * KE), 256, 0, stream>>>(pvpF, pvpT, Nn, D, KE);
  gemm(edgeV0A, pvpT, nullptr, vpF,   Nn, D, KE, D);
  gemm(edgeS0A, pvpT, nullptr, exvpF, Nn, D, KE, D);
  k_add_conv<<<blocks((size_t)Nn * D), 256, 0, stream>>>(vpF, exvpF, sumvpF, sumvpA,
                                                         (size_t)Nn * D);
  gemm(sumvpA, vnWt, vnb, tF, Nn, D, D, D);
  k_rownorm<<<Nn, 256, 0, stream>>>(tF, vpF, vprotF, vprotA, D, 1);

  // ---- UpdateVisualEdge (heavy, fused A-gen + async LDS-staged B) ----
  k_edge_gemm<<<(40000 + 127) / 128, 256, 0, stream>>>(vprotF, veW1t, veb1, G);
  k_edge_row<<<40000, 128, 0, stream>>>(G, veW2, veb2, edgeVF, preE);
  k_rowsoftmax<<<Nn, 256, 0, stream>>>(preE, edgeV1F, edgeV1A, Nn, KE, 0.1f);

  // ---- UpdateSemanticNode ----
  k_transpose<<<blocks((size_t)D * KE), 256, 0, stream>>>(sprotoF, spT, Nn, D, KE);
  gemm(edgeS0A, spT, nullptr, spFn,  Nn, D, KE, D);
  gemm(edgeV1A, spT, nullptr, exspF, Nn, D, KE, D);
  k_add_conv<<<blocks((size_t)Nn * D), 256, 0, stream>>>(spFn, exspF, sumspF, sumspA,
                                                         (size_t)Nn * D);
  gemm(sumspA, snWt, snb, t2F, Nn, D, D, D);
  k_rownorm<<<Nn, 256, 0, stream>>>(t2F, spFn, sprotF, sprotA, D, 1);

  // (UpdateSemanticEdge is dead code in the reference: its output is unused.)

  // ---- FusionLayer ----
  gemm(vprotA, fusWt, nullptr, Vv, Nn, D, D, D);
  gemm(sprotA, fusWt, nullptr, Vs, Nn, D, D, D);
  k_fusion<<<Nn, 256, 0, stream>>>(Vv, Vs, fusu, vprotF, sprotF, protoA);

  // ---- prob = img_feat @ proto_feat^T ----
  gemm(imgA, protoA, nullptr, out, Bb, Nn, D, Nn);
}
